// FeatureMapAttention_80891414053385
// MI455X (gfx1250) — matrix operand order repaired and checked by reading, not compiled, not run on hardware
//
#include <hip/hip_runtime.h>
#include <hip/hip_bf16.h>

// ---------------------------------------------------------------------------
// FeatureMapAttention on MI455X (gfx1250, wave32, WMMA).
//
// Math: q/k/v = W*f + b (1x1 conv).  S[m,n] = k_m . q_n / W,
// softmax over n (reference's softmax over axis=1), out = gamma * (V P) / L + v.
// => flash attention with Q:=k-projection, K:=q-projection, V:=v-projection.
// N = 8192, head dim C = 64, batch 4.  Never materialize the 1 GiB NxN matrix.
// Both matmuls use v_wmma_f32_16x16x32_f16 (f16 operands, f32 accumulate).
// Tile staging uses double-buffered GLOBAL_LOAD_ASYNC_TO_LDS_B128 (ASYNCcnt),
// so HBM/L2 latency overlaps the WMMA + softmax work.
// ---------------------------------------------------------------------------

typedef _Float16 half_t;
typedef __attribute__((ext_vector_type(16))) _Float16 v16h;
typedef __attribute__((ext_vector_type(8)))  float    v8f;

#define BB 4
#define CCH 64
#define NN 8192            // H*W = 64*128
#define SCALE (1.0f/128.0f)

#define MBLK 64            // m positions per block: 4 waves x 16
#define NSTEP 32           // n positions consumed per iteration (WMMA K for AV)
#define NITER (NN / NSTEP) // 256

#define QSTR 72            // LDS row strides (halfs), padded, 16B-aligned rows
#define VSTR 40
#define PSTR 40
#define QBUF (NSTEP * QSTR)   // 2304 halfs per buffer
#define VBUF (CCH * VSTR)     // 2560 halfs per buffer

__device__ __forceinline__ v8f wmma_f16(v16h a, v16h b, v8f c) {
    // D = A(16x32 f16) x B(32x16 f16) + C(16x16 f32)
    return __builtin_amdgcn_wmma_f32_16x16x32_f16(
        /*neg_a=*/false, a, /*neg_b=*/false, b,
        /*c_mod=*/(short)0, c, /*reuse_a=*/false, /*reuse_b=*/false);
}

// Build a 16-half fragment from two 16B-aligned chunks (elements 0..7, 8..15).
__device__ __forceinline__ v16h ld_frag(const half_t* p0, const half_t* p1) {
    union { v16h v; uint4 u[2]; } t;
    t.u[0] = *(const uint4*)p0;
    t.u[1] = *(const uint4*)p1;
    return t.v;
}

// Async copy of 16 bytes: global -> LDS, tracked by ASYNCcnt (per-lane).
// VDST carries the LDS byte address (low 32 bits of the generic LDS pointer).
__device__ __forceinline__ void async_ld16(half_t* lds, const half_t* g) {
    uint32_t l = (uint32_t)(uintptr_t)lds;
    uint64_t a = (uint64_t)(uintptr_t)g;
    asm volatile("global_load_async_to_lds_b128 %0, %1, off"
                 :: "v"(l), "v"(a) : "memory");
}
__device__ __forceinline__ void wait_async_le(int n) {
    if (n == 0) asm volatile("s_wait_asynccnt 0x0" ::: "memory");
    else        asm volatile("s_wait_asynccnt 0x4" ::: "memory");
}

// ---------------------------------------------------------------------------
// Kernel 1: projections q,k,v = W f + b.
//   qp, kp : f16 position-major  [B][N][C]   (feeds WMMA score fragments)
//   vp     : f16 channel-major   [B][C][N]   (A-matrix of AV matmul)
//   vf     : f32 channel-major   [B][C][N]   (residual: out = gamma*av + v)
// ---------------------------------------------------------------------------
__global__ void __launch_bounds__(256)
proj_kernel(const float* __restrict__ fq, const float* __restrict__ fk,
            const float* __restrict__ fv,
            const float* __restrict__ Wq, const float* __restrict__ bq,
            const float* __restrict__ Wk, const float* __restrict__ bk,
            const float* __restrict__ Wv, const float* __restrict__ bv,
            half_t* __restrict__ qp, half_t* __restrict__ kp,
            half_t* __restrict__ vp, float* __restrict__ vf)
{
    const int n = blockIdx.x * 256 + threadIdx.x;   // position, coalesced
    const int o = blockIdx.y;                       // output channel (scalar W reads)
    const int b = blockIdx.z;

    float aq = bq[o], ak = bk[o], av = bv[o];
    const size_t base = ((size_t)b * CCH) * NN + n;
    #pragma unroll 8
    for (int c = 0; c < CCH; ++c) {
        const size_t fi = base + (size_t)c * NN;
        aq = fmaf(Wq[o * CCH + c], fq[fi], aq);
        ak = fmaf(Wk[o * CCH + c], fk[fi], ak);
        av = fmaf(Wv[o * CCH + c], fv[fi], av);
    }
    const size_t po = ((size_t)(b * NN + n)) * CCH + o;
    qp[po] = (half_t)aq;
    kp[po] = (half_t)ak;
    const size_t vo = ((size_t)(b * CCH + o)) * NN + n;
    vp[vo] = (half_t)av;
    vf[vo] = av;
}

// ---------------------------------------------------------------------------
// Kernel 2: fused flash attention.
// Block = 128 threads (4 waves); each wave owns a 16-wide m-tile.
// Per n-step (32): 4 score WMMAs, online softmax, 4 AV WMMAs.
// D-fragment layout (ISA 7.12.2): m = vgpr + 8*(lane>=16), n = lane%16.
// ---------------------------------------------------------------------------
__global__ void __launch_bounds__(128)
flash_kernel(const half_t* __restrict__ qp, const half_t* __restrict__ kp,
             const half_t* __restrict__ vp, const float* __restrict__ vf,
             const float* __restrict__ gamma_p, float* __restrict__ out)
{
    __shared__ __align__(16) half_t sQ[2][QBUF];          // [n][c], c contiguous
    __shared__ __align__(16) half_t sV[2][VBUF];          // [c][n], n contiguous
    __shared__ __align__(16) half_t sP[4][16 * PSTR];     // per wave: [m][n]
    __shared__ float sStat[4][16];                        // per wave alpha/L bcast

    const int tid  = threadIdx.x;
    const int lane = tid & 31;
    const int wave = tid >> 5;
    const int l16  = lane & 15;
    const int hi   = lane >> 4;

    const int nbm  = NN / MBLK;                 // 128 m-blocks per batch
    const int bb   = blockIdx.x / nbm;
    const int mblk = blockIdx.x % nbm;
    const int m0   = mblk * MBLK + wave * 16;   // this wave's m-tile base

    const half_t* qbatch = qp + (size_t)bb * NN * CCH;
    const half_t* vbatch = vp + (size_t)bb * CCH * NN;

    // Per-thread staging coordinates (fixed): 4 async b128 per thread per buffer.
    const int qr = tid >> 2, qc = (tid & 3) * 16;        // q tile: 32 x 64 halfs
    const int vr = tid >> 1, vc = (tid & 1) * 16;        // v tile: 64 x 32 halfs

    // Issue async staging of one n-step tile into buffer `buf`.
    auto prefetch = [&](int buf, int nb) {
        const half_t* qg = qbatch + (size_t)(nb + qr) * CCH + qc;
        half_t*       qd = &sQ[buf][qr * QSTR + qc];
        async_ld16(qd,     qg);
        async_ld16(qd + 8, qg + 8);
        const half_t* vg = vbatch + (size_t)vr * NN + nb + vc;
        half_t*       vd = &sV[buf][vr * VSTR + vc];
        async_ld16(vd,     vg);
        async_ld16(vd + 8, vg + 8);
    };

    // Persistent A-fragments of the score matmul: rows = m, K = c (two 32-chunks).
    // A layout: halves h -> K = c0 + h (contiguous 16), c0 = 16*hi (+32 for chunk 1).
    const half_t* krow = kp + ((size_t)(bb * NN + m0 + l16)) * CCH;
    const int a0 = 16 * hi;
    const v16h kf0 = ld_frag(krow + a0,      krow + a0 + 8);
    const v16h kf1 = ld_frag(krow + a0 + 32, krow + a0 + 40);

    v8f acc[4];                                 // av tile: c in 4 chunks of 16, x m(16)
    #pragma unroll
    for (int i = 0; i < 4; ++i)
        #pragma unroll
        for (int j = 0; j < 8; ++j) acc[i][j] = 0.0f;

    float mrun[8], lrun[8];                     // per-lane stats for m = j + 8*hi
    #pragma unroll
    for (int j = 0; j < 8; ++j) { mrun[j] = -3.0e38f; lrun[j] = 0.0f; }

    prefetch(0, 0);                             // prologue

    for (int it = 0; it < NITER; ++it) {
        const int cur = it & 1;
        // Issue next buffer (wraps on the last iteration -> uniform wait count).
        prefetch(cur ^ 1, ((it + 1) & (NITER - 1)) * NSTEP);
        // Our 4 newest async ops may be outstanding; the previous 4 (current
        // buffer) are complete once ASYNCcnt <= 4 (async loads retire in order).
        wait_async_le(4);
        __syncthreads();                        // cross-wave visibility of tiles

        const half_t* q0r = &sQ[cur][(0 * 16 + l16) * QSTR];
        const half_t* q1r = &sQ[cur][(1 * 16 + l16) * QSTR];
        const half_t* vb  = &sV[cur][0];

        // ---- scores: S[m, n] for two 16-wide n subtiles ----
        v8f s0 = {}, s1 = {};
        {
            // B layout: lane holds column n = lane%16, halves h -> K = h + 16*hi.
            const int b0 = 16 * hi;
            v16h qb;
            qb = ld_frag(q0r + b0,      q0r + b0 + 8);      s0 = wmma_f16(kf0, qb, s0);
            qb = ld_frag(q0r + 32 + b0, q0r + 32 + b0 + 8); s0 = wmma_f16(kf1, qb, s0);
            qb = ld_frag(q1r + b0,      q1r + b0 + 8);      s1 = wmma_f16(kf0, qb, s1);
            qb = ld_frag(q1r + 32 + b0, q1r + 32 + b0 + 8); s1 = wmma_f16(kf1, qb, s1);
        }

        // ---- online softmax over n (reduce across the 16-lane half-groups) ----
        float p0[8], p1[8], alpha[8];
        #pragma unroll
        for (int j = 0; j < 8; ++j) {
            float a = s0[j] * SCALE;
            float b = s1[j] * SCALE;
            float t = fmaxf(a, b);
            t = fmaxf(t, __shfl_xor(t, 1));
            t = fmaxf(t, __shfl_xor(t, 2));
            t = fmaxf(t, __shfl_xor(t, 4));
            t = fmaxf(t, __shfl_xor(t, 8));
            const float mn = fmaxf(mrun[j], t);
            alpha[j] = __expf(mrun[j] - mn);
            mrun[j]  = mn;
            p0[j] = __expf(a - mn);
            p1[j] = __expf(b - mn);
            float r = p0[j] + p1[j];
            r += __shfl_xor(r, 1);
            r += __shfl_xor(r, 2);
            r += __shfl_xor(r, 4);
            r += __shfl_xor(r, 8);
            lrun[j] = lrun[j] * alpha[j] + r;
        }

        // ---- broadcast alpha[m] to the accumulator layout (m = lane%16) ----
        if (l16 == 0) {
            #pragma unroll
            for (int j = 0; j < 8; ++j) sStat[wave][8 * hi + j] = alpha[j];
        }
        const float am = sStat[wave][l16];
        #pragma unroll
        for (int c4 = 0; c4 < 4; ++c4)
            #pragma unroll
            for (int j = 0; j < 8; ++j) acc[c4][j] *= am;

        // ---- P tile to LDS in B-fragment layout: [m][n], n contiguous ----
        half_t* pw = sP[wave];
        #pragma unroll
        for (int j = 0; j < 8; ++j) {
            pw[(j + 8 * hi) * PSTR + l16     ] = (half_t)p0[j];
            pw[(j + 8 * hi) * PSTR + l16 + 16] = (half_t)p1[j];
        }

        // ---- AV: acc[c, m] += V(16c x 32n) x P(32n x 16m), 4 c-chunks ----
        const v16h pb = ld_frag(pw + l16 * PSTR + 16 * hi,
                                pw + l16 * PSTR + 16 * hi + 8);
        #pragma unroll
        for (int c4 = 0; c4 < 4; ++c4) {
            const half_t* vrow = vb + (size_t)(c4 * 16 + l16) * VSTR;
            const v16h va = ld_frag(vrow + 16 * hi, vrow + 16 * hi + 8);
            acc[c4] = wmma_f16(va, pb, acc[c4]);
        }
        __syncthreads();                        // tiles fully consumed
    }

    // ---- finalize: out = gamma * acc / L + v ----
    if (l16 == 0) {
        #pragma unroll
        for (int j = 0; j < 8; ++j) sStat[wave][8 * hi + j] = lrun[j];
    }
    const float invL = 1.0f / sStat[wave][l16];
    const float g = gamma_p[0];
    #pragma unroll
    for (int c4 = 0; c4 < 4; ++c4)
        #pragma unroll
        for (int j = 0; j < 8; ++j) {
            const int c = c4 * 16 + j + 8 * hi;
            const size_t o = ((size_t)(bb * CCH + c)) * NN + m0 + l16;
            out[o] = g * acc[c4][j] * invL + vf[o];
        }
}

// ---------------------------------------------------------------------------
extern "C" void kernel_launch(void* const* d_in, const int* in_sizes, int n_in,
                              void* d_out, int out_size, void* d_ws, size_t ws_size,
                              hipStream_t stream) {
    (void)in_sizes; (void)n_in; (void)out_size; (void)ws_size;
    const float* fq = (const float*)d_in[0];
    const float* fk = (const float*)d_in[1];
    const float* fv = (const float*)d_in[2];
    const float* Wq = (const float*)d_in[3];
    const float* bq = (const float*)d_in[4];
    const float* Wk = (const float*)d_in[5];
    const float* bk = (const float*)d_in[6];
    const float* Wv = (const float*)d_in[7];
    const float* bv = (const float*)d_in[8];
    const float* gm = (const float*)d_in[9];

    // Workspace layout (20 MB total):
    //   qp f16 [B][N][C] @ 0        (4 MB)
    //   kp f16 [B][N][C] @ 4 MB     (4 MB)
    //   vp f16 [B][C][N] @ 8 MB     (4 MB)
    //   vf f32 [B][C][N] @ 12 MB    (8 MB)
    char* ws = (char*)d_ws;
    half_t* qp = (half_t*)(ws);
    half_t* kp = (half_t*)(ws + (4u << 20));
    half_t* vp = (half_t*)(ws + (8u << 20));
    float*  vf = (float*)(ws + (12u << 20));
    float*  out = (float*)d_out;

    dim3 g1(NN / 256, CCH, BB);
    proj_kernel<<<g1, 256, 0, stream>>>(fq, fk, fv, Wq, bq, Wk, bk, Wv, bv,
                                        qp, kp, vp, vf);

    dim3 g2(BB * (NN / MBLK));      // 512 blocks x 128 threads (4 waves)
    flash_kernel<<<g2, 128, 0, stream>>>(qp, kp, vp, vf, gm, out);
}
